// DroneRelationModel_51616916963853
// MI455X (gfx1250) — matrix operand order repaired and checked by reading, not compiled, not run on hardware
//
#include <hip/hip_runtime.h>
#include <hip/hip_bf16.h>
#include <math.h>

typedef __attribute__((ext_vector_type(16))) __bf16 v16bf;
typedef __attribute__((ext_vector_type(8)))  __bf16 v8bf;
typedef __attribute__((ext_vector_type(8)))  float  v8f;
typedef __attribute__((ext_vector_type(4)))  float  v4f;

#define HDIM     512
#define KDIM     2048          // 4*H concatenated features
#define BM       32            // pairs per block (2 M-tiles of 16)
#define KT       32            // K chunk per WMMA step
#define NTHREADS 256           // 8 waves (wave32)

__device__ __forceinline__ unsigned short f2bf(float f) {
    unsigned int u = __builtin_bit_cast(unsigned int, f);
    u += 0x7fffu + ((u >> 16) & 1u);          // round-to-nearest-even
    return (unsigned short)(u >> 16);
}
__device__ __forceinline__ unsigned int pack2(float a, float b) {
    return (unsigned int)f2bf(a) | ((unsigned int)f2bf(b) << 16);
}

// ---- One-time prep: Wt[n][k] = bf16(W1[k][n])  (2 MB in d_ws) ----
__global__ __launch_bounds__(256)
void w1_to_bf16_t_kernel(const float* __restrict__ W1, unsigned short* __restrict__ Wt)
{
    int idx = blockIdx.x * 256 + threadIdx.x;          // over KDIM*HDIM/4 tasks
    if (idx >= (KDIM * HDIM) / 4) return;
    int k  = idx >> 7;                                 // / (HDIM/4)
    int n4 = (idx & 127) << 2;
    v4f w = *(const v4f*)(W1 + (long)k * HDIM + n4);
    Wt[(long)(n4 + 0) * KDIM + k] = f2bf(w[0]);
    Wt[(long)(n4 + 1) * KDIM + k] = f2bf(w[1]);
    Wt[(long)(n4 + 2) * KDIM + k] = f2bf(w[2]);
    Wt[(long)(n4 + 3) * KDIM + k] = f2bf(w[3]);
}

template <bool USE_WT>
__global__ __launch_bounds__(NTHREADS)
void drone_pair_mlp_kernel(const float* __restrict__ ctx,           // [NUM_DRONES, 512]
                           const int*   __restrict__ rel,           // [NUM_PAIRS, 2]
                           const float* __restrict__ W1,            // [2048, 512] f32 (fallback)
                           const unsigned short* __restrict__ Wt,   // [512, 2048] bf16 (prepped)
                           const float* __restrict__ b1,
                           const float* __restrict__ W2,
                           const float* __restrict__ b2,
                           float* __restrict__ out,
                           int npairs)
{
    __shared__ unsigned short As[BM][KDIM];   // 128 KB, bf16 pair features
    __shared__ unsigned short Bs[HDIM][KT];   //  32 KB, W1 chunk, [n][k]
    // 160 KB total -> 2 workgroups per 320 KB WGP

    const int tid  = threadIdx.x;
    const int lane = tid & 31;
    const int wave = tid >> 5;       // each wave owns N tiles [wave*4, wave*4+3]
    const int half = lane >> 4;
    const int l16  = lane & 15;
    const int p0   = blockIdx.x * BM;

    // ---- Stage A: gather src/dst, build [s, d, d-s, s*d], convert to bf16 ----
    for (int t = tid; t < BM * (HDIM / 4); t += NTHREADS) {
        int r  = t >> 7;
        int c4 = (t & 127) << 2;
        int p  = p0 + r; if (p >= npairs) p = npairs - 1;
        int si = rel[2 * p + 0];
        int di = rel[2 * p + 1];
        v4f s = *(const v4f*)(ctx + (long)si * HDIM + c4);
        v4f d = *(const v4f*)(ctx + (long)di * HDIM + c4);
        *(uint2*)&As[r][0 * HDIM + c4] = make_uint2(pack2(s[0],      s[1]),      pack2(s[2],      s[3]));
        *(uint2*)&As[r][1 * HDIM + c4] = make_uint2(pack2(d[0],      d[1]),      pack2(d[2],      d[3]));
        *(uint2*)&As[r][2 * HDIM + c4] = make_uint2(pack2(d[0]-s[0], d[1]-s[1]), pack2(d[2]-s[2], d[3]-s[3]));
        *(uint2*)&As[r][3 * HDIM + c4] = make_uint2(pack2(s[0]*d[0], s[1]*d[1]), pack2(s[2]*d[2], s[3]*d[3]));
    }

    const unsigned lds_bs = (unsigned)(size_t)&Bs[0][0];   // LDS offset = low 32 bits
    v8f acc[2][4] = {};

    for (int kt = 0; kt < KDIM; kt += KT) {
        __syncthreads();   // prior-iteration Bs readers done; also covers As staging

        if (USE_WT) {
            // ---- Async DMA: 64 contiguous bytes per row of Wt into Bs[n][0..31] ----
            #pragma unroll
            for (int i = 0; i < (KT * HDIM * 2) / (16 * NTHREADS); ++i) {   // 8
                int idx = i * NTHREADS + tid;            // 0..2047, 16 B each
                int n   = idx >> 2;
                int seg = idx & 3;
                unsigned goff  = (unsigned)(n * (KDIM * 2) + kt * 2 + seg * 16);
                unsigned laddr = lds_bs + (unsigned)(n * (KT * 2) + seg * 16);
                asm volatile("global_load_async_to_lds_b128 %0, %1, %2"
                             :: "v"(laddr), "v"(goff), "s"(Wt) : "memory");
            }
            if (kt + KT < KDIM) {                        // warm next chunk (512 rows)
                const unsigned short* nxt = Wt + (long)(2 * tid) * KDIM + (kt + KT);
                __builtin_prefetch(nxt, 0, 1);
                __builtin_prefetch(nxt + KDIM, 0, 1);
            }
            asm volatile("s_wait_asynccnt 0" ::: "memory");
        } else {
            // ---- Fallback: stage from f32 W1, converting + transposing ----
            for (int t = tid; t < (KT * HDIM) / 4; t += NTHREADS) {
                int kk = t >> 7;
                int n4 = (t & 127) << 2;
                v4f w = *(const v4f*)(W1 + (long)(kt + kk) * HDIM + n4);
                Bs[n4 + 0][kk] = f2bf(w[0]);
                Bs[n4 + 1][kk] = f2bf(w[1]);
                Bs[n4 + 2][kk] = f2bf(w[2]);
                Bs[n4 + 3][kk] = f2bf(w[3]);
            }
            if (kt + KT < KDIM) {
                const float* nxt = W1 + (long)(kt + KT) * HDIM;
                __builtin_prefetch(nxt + tid * 64, 0, 1);
                __builtin_prefetch(nxt + tid * 64 + 32, 0, 1);
            }
        }
        __syncthreads();

        // ---- A fragments for both M-tiles (16-bit A layout) ----
        v16bf a[2];
        #pragma unroll
        for (int mt = 0; mt < 2; ++mt) {
            int row = mt * 16 + l16;
            v8bf x0 = *(const v8bf*)&As[row][kt + half * 8];
            v8bf x1 = *(const v8bf*)&As[row][kt + 16 + half * 8];
            #pragma unroll
            for (int i = 0; i < 8; ++i) { a[mt][i] = x0[i]; a[mt][8 + i] = x1[i]; }
        }

        #pragma unroll
        for (int t = 0; t < 4; ++t) {
            int n = (wave * 4 + t) * 16 + l16;
            v16bf bfrag;
            v8bf y0 = *(const v8bf*)&Bs[n][half * 8];
            v8bf y1 = *(const v8bf*)&Bs[n][16 + half * 8];
            #pragma unroll
            for (int i = 0; i < 8; ++i) { bfrag[i] = y0[i]; bfrag[8 + i] = y1[i]; }

            acc[0][t] = __builtin_amdgcn_wmma_f32_16x16x32_bf16(
                false, a[0], false, bfrag, (short)0, acc[0][t], false, false);
            acc[1][t] = __builtin_amdgcn_wmma_f32_16x16x32_bf16(
                false, a[1], false, bfrag, (short)0, acc[1][t], false, false);
        }
    }

    // ---- Epilogue: bias + ReLU, dot with W2, reduce over N ----
    __syncthreads();                       // all fragment reads of As complete
    float* partial = (float*)&As[0][0];    // alias dead As as reduction buffer
    if (tid < BM) partial[tid] = 0.0f;
    __syncthreads();

    #pragma unroll
    for (int mt = 0; mt < 2; ++mt) {
        #pragma unroll
        for (int v = 0; v < 8; ++v) {
            float contrib = 0.0f;
            #pragma unroll
            for (int t = 0; t < 4; ++t) {
                int n  = (wave * 4 + t) * 16 + l16;
                float hv = acc[mt][t][v] + b1[n];
                hv = fmaxf(hv, 0.0f);
                contrib += hv * W2[n];
            }
            contrib += __shfl_xor(contrib, 1);
            contrib += __shfl_xor(contrib, 2);
            contrib += __shfl_xor(contrib, 4);
            contrib += __shfl_xor(contrib, 8);
            if (l16 == 0) atomicAdd(&partial[mt * 16 + v + 8 * half], contrib);
        }
    }
    __syncthreads();

    if (tid < BM && (p0 + tid) < npairs) {
        float z = partial[tid] + b2[0];
        out[p0 + tid] = 1.0f / (1.0f + __expf(-z));
    }
}

extern "C" void kernel_launch(void* const* d_in, const int* in_sizes, int n_in,
                              void* d_out, int out_size, void* d_ws, size_t ws_size,
                              hipStream_t stream) {
    // setup_inputs order: current_features(0, unused), context_embeddings(1),
    // relationships(2), W1(3), b1(4), W2(5), b2(6)
    const float* ctx = (const float*)d_in[1];
    const int*   rel = (const int*)  d_in[2];
    const float* W1  = (const float*)d_in[3];
    const float* b1  = (const float*)d_in[4];
    const float* W2  = (const float*)d_in[5];
    const float* b2  = (const float*)d_in[6];
    float* out = (float*)d_out;

    int npairs = in_sizes[2] / 2;            // 131072
    int grid   = (npairs + BM - 1) / BM;     // 4096 blocks

    const size_t wt_bytes = (size_t)HDIM * KDIM * sizeof(unsigned short);  // 2 MB
    if (ws_size >= wt_bytes) {
        unsigned short* Wt = (unsigned short*)d_ws;
        w1_to_bf16_t_kernel<<<(KDIM * HDIM / 4 + 255) / 256, 256, 0, stream>>>(W1, Wt);
        drone_pair_mlp_kernel<true><<<grid, NTHREADS, 0, stream>>>(
            ctx, rel, W1, Wt, b1, W2, b2, out, npairs);
    } else {
        drone_pair_mlp_kernel<false><<<grid, NTHREADS, 0, stream>>>(
            ctx, rel, W1, nullptr, b1, W2, b2, out, npairs);
    }
}
